// GraphAttentionLayer_31645319037629
// MI455X (gfx1250) — matrix operand order repaired and checked by reading, not compiled, not run on hardware
//
#include <hip/hip_runtime.h>
#include <hip/hip_bf16.h>

#define GAT_N 12288
#define GAT_D 256
#define GAT_F 64
#define GAT_LOG2E 1.44269504088896340736f

typedef __attribute__((ext_vector_type(16))) _Float16 v16h;
typedef __attribute__((ext_vector_type(8)))  _Float16 v8h;
typedef __attribute__((ext_vector_type(8)))  float    v8f;
typedef __attribute__((ext_vector_type(4)))  float    v4f;
typedef __attribute__((ext_vector_type(4)))  int      v4i;

// ---------------------------------------------------------------------------
// Kernel 0: init the w2-max cell (float max via signed-int bit trick; max of
// 12288 ~N(0,180) samples is certainly > 0, so signed atomicMax on raw bits
// yields the correct float max).
// ---------------------------------------------------------------------------
__global__ void gat_init_kernel(int* w2max_bits) {
    *w2max_bits = (int)0x80000000; // INT_MIN: below every float bit pattern
}

// ---------------------------------------------------------------------------
// Kernel 1: h = x @ W (f32 accum), hT = h^T in f16, and log2-scaled scores
// w1L = (h@a_src)*log2e, w2L = (h@a_dst)*log2e, plus global max of w2L.
// 0.4 GFLOP — negligible; W staged in LDS.
// ---------------------------------------------------------------------------
__global__ __launch_bounds__(128) void gat_h_kernel(
    const float* __restrict__ x, const float* __restrict__ W,
    const float* __restrict__ am, float* __restrict__ w1L,
    float* __restrict__ w2L, _Float16* __restrict__ hT,
    int* __restrict__ w2max_bits)
{
    __shared__ float Ws[GAT_D * GAT_F];   // 64 KB
    __shared__ float As[2 * GAT_F];

    const int tid = threadIdx.x;
    for (int i = tid; i < GAT_D * GAT_F; i += 128) Ws[i] = W[i];
    if (tid < 2 * GAT_F) As[tid] = am[tid];
    __syncthreads();

    const int row = blockIdx.x * 128 + tid;
    const float* xr = x + (size_t)row * GAT_D;

    float acc[GAT_F];
#pragma unroll
    for (int c = 0; c < GAT_F; ++c) acc[c] = 0.0f;

    for (int k = 0; k < GAT_D; ++k) {
        const float xk = xr[k];
        const float* wk = &Ws[k * GAT_F];
#pragma unroll
        for (int c = 0; c < GAT_F; ++c) acc[c] = fmaf(xk, wk[c], acc[c]);
    }

    float s1 = 0.0f, s2 = 0.0f;
#pragma unroll
    for (int c = 0; c < GAT_F; ++c) {
        s1 = fmaf(acc[c], As[c], s1);
        s2 = fmaf(acc[c], As[GAT_F + c], s2);
        hT[(size_t)c * GAT_N + row] = (_Float16)acc[c];   // transposed f16 copy
    }
    const float s2s = s2 * GAT_LOG2E;
    w1L[row] = s1 * GAT_LOG2E;
    w2L[row] = s2s;
    atomicMax(w2max_bits, __float_as_int(s2s));
}

// ---------------------------------------------------------------------------
// Kernel 2: fused masked-softmax attention + (attention @ h), single pass
// over adj. One wave32 per 16-row output tile. Per 32-col step:
//   - A = 16x32 f16 probability tile, built in the WMMA A register layout
//     (exact softmax: exp2 of log2-scaled scores shifted by a safe per-row
//     upper bound derived from the *global* w2 max — monotonicity of lrelu
//     makes it dominate every masked entry, so no online rescaling needed)
//   - 4x v_wmma_f32_16x16x32_f16 accumulate the 16x64 numerator
//   - 1x extra WMMA against an all-ones B computes the row sums (softmax
//     denominator) directly in the C layout -> no cross-lane reduction at all
// ---------------------------------------------------------------------------
__global__ __launch_bounds__(32) void gat_attn_kernel(
    const int* __restrict__ adj, const float* __restrict__ w1L,
    const float* __restrict__ w2L, const _Float16* __restrict__ hT,
    const int* __restrict__ w2max_bits, float* __restrict__ out)
{
    const int lane = threadIdx.x & 31;
    const int g    = lane >> 4;      // lane group (k-half selector)
    const int n    = lane & 15;      // A-row for this lane / B,C column
    const int rowbase = blockIdx.x * 16;
    const int row  = rowbase + n;    // row whose A elements this lane builds

    const float w2maxL = __int_as_float(*w2max_bits);
    const float w1r    = w1L[row];
    const float tmax   = w1r + w2maxL;
    const float miL    = tmax > 0.0f ? tmax : 0.01f * tmax;  // safe upper bound

    const int* adjRow = adj + (size_t)row * GAT_N;

    v8f c[4];       // numerator tiles (cols t*16+n, rows r+8g)
    v8f csum;       // denominator tile (every column = row sums)
#pragma unroll
    for (int t = 0; t < 4; ++t) c[t] = (v8f){0.f,0.f,0.f,0.f,0.f,0.f,0.f,0.f};
    csum = (v8f){0.f,0.f,0.f,0.f,0.f,0.f,0.f,0.f};

    v16h bones;
#pragma unroll
    for (int e = 0; e < 16; ++e) bones[e] = (_Float16)1.0f;

#pragma unroll 2
    for (int jb = 0; jb < GAT_N; jb += 32) {
        // prefetch adj 8 steps (1 KB) ahead -> global_prefetch_b8
        __builtin_prefetch(adjRow + jb + 256, 0, 0);

        // A-layout: element e (0..15) of this lane covers
        //   k = 8*g + e          for e in [0,8)
        //   k = 16 + 8*g + (e-8) for e in [8,16)
        // => two contiguous 8-int runs per lane.
        const v4i q0 = *(const v4i*)(adjRow + jb + 8 * g);
        const v4i q1 = *(const v4i*)(adjRow + jb + 8 * g + 4);
        const v4i q2 = *(const v4i*)(adjRow + jb + 8 * g + 16);
        const v4i q3 = *(const v4i*)(adjRow + jb + 8 * g + 20);
        const v4f u0 = *(const v4f*)(w2L + jb + 8 * g);
        const v4f u1 = *(const v4f*)(w2L + jb + 8 * g + 4);
        const v4f u2 = *(const v4f*)(w2L + jb + 8 * g + 16);
        const v4f u3 = *(const v4f*)(w2L + jb + 8 * g + 20);

        v16h a;
#pragma unroll
        for (int e = 0; e < 16; ++e) {
            int   adjv;
            float w2v;
            if      (e <  4) { adjv = q0[e];      w2v = u0[e];      }
            else if (e <  8) { adjv = q1[e -  4]; w2v = u1[e -  4]; }
            else if (e < 12) { adjv = q2[e -  8]; w2v = u2[e -  8]; }
            else             { adjv = q3[e - 12]; w2v = u3[e - 12]; }
            const float s  = w1r + w2v;                      // log2-scaled score
            const float ev = s > 0.0f ? s : 0.01f * s;       // leaky_relu(0.01)
            const float p  = (adjv > 0)
                           ? __builtin_amdgcn_exp2f(ev - miL) // exact exp(e - m)
                           : 0.0f;
            a[e] = (_Float16)p;
        }

        // B tiles: hT is [64][N] f16, so each lane's 16 B-layout elements
        // (k = 8*g + e for e<8, k = 16 + 8*g + (e-8) for e>=8, n fixed) are two 16-byte runs (L2-resident).
#pragma unroll
        for (int t = 0; t < 4; ++t) {
            const _Float16* bp = hT + (size_t)(t * 16 + n) * GAT_N + jb + 8 * g;
            const v8h lo = *(const v8h*)bp;
            const v8h hi = *(const v8h*)(bp + 16);
            const v16h b = __builtin_shufflevector(lo, hi,
                0, 1, 2, 3, 4, 5, 6, 7, 8, 9, 10, 11, 12, 13, 14, 15);
            c[t] = __builtin_amdgcn_wmma_f32_16x16x32_f16(
                false, a, false, b, (short)0, c[t], false, false);
        }
        // Row sums on the matrix unit: csum rows land in the same C layout.
        csum = __builtin_amdgcn_wmma_f32_16x16x32_f16(
            false, a, false, bones, (short)0, csum, false, false);
    }

    // Normalize + final leaky_relu(0.2). csum[r] already holds the full
    // denominator for row (r + 8g) in every column — no shuffles needed.
#pragma unroll
    for (int r = 0; r < 8; ++r) {
        const float invr = 1.0f / csum[r];
        float* orow = out + (size_t)(rowbase + r + 8 * g) * GAT_F + n;
#pragma unroll
        for (int t = 0; t < 4; ++t) {
            float v = c[t][r] * invr;
            v = v > 0.0f ? v : 0.2f * v;               // leaky_relu(0.2)
            orow[t * 16] = v;
        }
    }
}

// ---------------------------------------------------------------------------
extern "C" void kernel_launch(void* const* d_in, const int* in_sizes, int n_in,
                              void* d_out, int out_size, void* d_ws, size_t ws_size,
                              hipStream_t stream) {
    const float* x   = (const float*)d_in[0];
    const int*   adj = (const int*)d_in[1];
    const float* W   = (const float*)d_in[2];
    const float* am  = (const float*)d_in[3];
    float* out = (float*)d_out;

    char* ws = (char*)d_ws;
    int*       w2max_bits = (int*)ws;
    float*     w1L = (float*)(ws + 64);
    float*     w2L = (float*)(ws + 64 + (size_t)GAT_N * 4);
    _Float16*  hT  = (_Float16*)(ws + 64 + 2 * (size_t)GAT_N * 4);
    // ws usage: 64 + 2*N*4 + 64*N*2 bytes ~= 1.67 MB

    gat_init_kernel<<<1, 1, 0, stream>>>(w2max_bits);
    gat_h_kernel<<<GAT_N / 128, 128, 0, stream>>>(x, W, am, w1L, w2L, hT, w2max_bits);
    gat_attn_kernel<<<GAT_N / 16, 32, 0, stream>>>(adj, w1L, w2L, hT, w2max_bits, out);
}
